// PatchFlow_76785425318058
// MI455X (gfx1250) — matrix operand order repaired and checked by reading, not compiled, not run on hardware
//
#include <hip/hip_runtime.h>
#include <hip/hip_bf16.h>

// ---------------------------------------------------------------------------
// Types for CDNA5 WMMA (wave32): bf16 A/B fragments (16 bf16/lane = 8 VGPRs),
// f32 C/D accumulator (8 f32/lane = 8 VGPRs).
// ---------------------------------------------------------------------------
typedef __attribute__((ext_vector_type(16))) __bf16 v16bf;
typedef __attribute__((ext_vector_type(8)))  __bf16 v8bf;
typedef __attribute__((ext_vector_type(8)))  float  v8f;

#define WMMA_BF16(A, B, C) \
  __builtin_amdgcn_wmma_f32_16x16x32_bf16(false, (A), false, (B), (short)0, (C), false, false)

// Problem constants (match reference).
#define BB      16
#define CIN     64
#define HIN     128
#define EE      128
#define PP      1024     // 32*32 pooled patches
#define MM      16384    // B*P rows
#define SPLIT   32
#define SUBIN   288      // 32 + 2*E
#define SUBHID  576

// ---------------------------------------------------------------------------
// 1) L2 pooling: pooled = sqrt(sum_{7x7, stride4, pad2} x^2).
//    Writes z0 in flow layout [m = p*16+b][64] (f32) and A1 in transformer
//    layout [b*1024+p][64] (bf16) for the first GEMM.
// ---------------------------------------------------------------------------
__global__ void __launch_bounds__(256) pool_kernel(const float* __restrict__ x,
                                                   float* __restrict__ z0,
                                                   __bf16* __restrict__ A1) {
  size_t i = (size_t)blockIdx.x * 256 + threadIdx.x;          // over 1024*16*64
  if (i >= (size_t)PP * BB * CIN) return;
  int c = i & 63;
  int rest = (int)(i >> 6);
  int batch = rest & 15;
  int p = rest >> 4;
  int py = p >> 5, px = p & 31;
  int y0 = py * 4 - 2, x0 = px * 4 - 2;
  const float* xb = x + (((size_t)batch * CIN + c) * 128) * 128;
  float s = 0.f;
  for (int ky = 0; ky < 7; ++ky) {
    int yy = y0 + ky;
    if (yy < 0 || yy >= 128) continue;
    for (int kx = 0; kx < 7; ++kx) {
      int xx = x0 + kx;
      if (xx < 0 || xx >= 128) continue;
      float t = xb[yy * 128 + xx];
      s += t * t;
    }
  }
  float r = sqrtf(s);
  z0[((size_t)p * BB + batch) * CIN + c] = r;
  A1[((size_t)batch * PP + p) * CIN + c] = (__bf16)r;
}

// ---------------------------------------------------------------------------
// 2) 2D positional encoding, pos[p][e], p = y*32 + x.
// ---------------------------------------------------------------------------
__global__ void __launch_bounds__(256) pos_kernel(float* __restrict__ pos) {
  int i = blockIdx.x * 256 + threadIdx.x;                     // over 1024*128
  if (i >= PP * EE) return;
  int p = i >> 7, e = i & 127;
  int yy = p >> 5, xx = p & 31;
  int hi = e >> 6;                // 0: x-based, 1: y-based
  int ee = e & 63;
  int j = ee >> 1;
  float div = __expf(-logf(10000.f) * (float)(2 * j) / 64.f);
  float arg = (hi ? (float)yy : (float)xx) * div;
  pos[i] = (ee & 1) ? cosf(arg) : sinf(arg);
}

// ---------------------------------------------------------------------------
// 3) fp32 -> bf16 weight conversion.
// ---------------------------------------------------------------------------
__global__ void __launch_bounds__(256) f2b(const float* __restrict__ in,
                                           __bf16* __restrict__ out, int n) {
  int i = blockIdx.x * 256 + threadIdx.x;
  if (i < n) out[i] = (__bf16)in[i];
}

// ---------------------------------------------------------------------------
// 4) Wave32 WMMA GEMM with LDS-staged weights:
//    C[M][N] = A[M][K] * W[N][K]^T (+bias)(+ReLU).
//    Each 256-thread block owns one 32-wide N panel and 8 M-tiles (128 rows).
//    The weight panel is copied ONCE into LDS with per-lane
//    global_load_async_to_lds_b128 (ASYNCcnt), pre-permuted into WMMA
//    B-fragment order, so every B operand is a single contiguous 32B ds read.
//    Fragment layouts used here (contiguous k, same order for A and B):
//      A (16x32 bf16): lane<16 -> row=lane,    K {k..k+15}
//                      lane>=16 -> row=lane-16, K {k+16..k+31}
//      B (32x16 bf16): col = lane%16, K half = (lane>=16)*16, 16 contiguous.
//      C (16x16 f32) : VGPR r -> row r + 8*(lane>=16), col = lane%16.
//    LDS image: slot[(ks*2+s)][lane] = 32B = the exact B fragment bytes.
// ---------------------------------------------------------------------------
__global__ void __launch_bounds__(256) gemm_bf16_wmma(
    const __bf16* __restrict__ A, const __bf16* __restrict__ W,
    const float* __restrict__ bias, float* __restrict__ Cf,
    __bf16* __restrict__ Cb, int M, int N, int K, int relu) {
  extern __shared__ __align__(128) __bf16 wsm[];   // (K/32)*2 slots * 32 lanes * 16 bf16
  const int lane = threadIdx.x & 31;
  const int wave = threadIdx.x >> 5;
  const int npanels = N >> 5;
  const int nt = blockIdx.x % npanels;
  const int mg = blockIdx.x / npanels;
  const int n0 = nt << 5;
  const int m0 = (mg * 8 + wave) << 4;
  const int hl = lane >> 4, l16 = lane & 15;

  // --- cooperative async copy: global W panel -> LDS in B-fragment order ---
  {
    const int total = (K >> 5) * 128;              // 16B chunks
    for (int ci = threadIdx.x; ci < total; ci += 256) {
      int ks = ci >> 7;                            // k-step (32 wide)
      int r  = ci & 127;
      int s  = (r >> 6) & 1;                       // N subtile (0/1)
      int l  = (r >> 1) & 31;                      // destination lane slot
      int c  = r & 1;                              // 16B chunk within 32B
      int lh = l >> 4, ll = l & 15;
      const __bf16* gp = W + (size_t)(n0 + s * 16 + ll) * K + ks * 32 + lh * 16 + c * 8;
      unsigned lda = (unsigned)(uintptr_t)(wsm + ((size_t)((ks * 2 + s) * 32 + l) * 16 + c * 8));
      unsigned long long ga = (unsigned long long)(uintptr_t)gp;
      asm volatile("global_load_async_to_lds_b128 %0, %1, off"
                   :: "v"(lda), "v"(ga) : "memory");
    }
    asm volatile("s_wait_asynccnt 0x0" ::: "memory");
    __syncthreads();
  }

  const __bf16* Ap = A + (size_t)(m0 + l16) * K + (hl ? 16 : 0);
  v8f acc0 = {}, acc1 = {};
  const int nks = K >> 5;
  for (int ks = 0; ks < nks; ++ks) {
    __builtin_prefetch(Ap + ks * 32 + 256, 0, 1);  // global_prefetch_b8
    v16bf af;
    ((v8bf*)&af)[0] = *(const v8bf*)(Ap + ks * 32);
    ((v8bf*)&af)[1] = *(const v8bf*)(Ap + ks * 32 + 8);
    v16bf b0 = *(const v16bf*)(wsm + ((size_t)(ks * 2 + 0) * 32 + lane) * 16);
    v16bf b1 = *(const v16bf*)(wsm + ((size_t)(ks * 2 + 1) * 32 + lane) * 16);
    acc0 = WMMA_BF16(af, b0, acc0);
    acc1 = WMMA_BF16(af, b1, acc1);
  }
  float bc0 = bias ? bias[n0 + l16] : 0.f;
  float bc1 = bias ? bias[n0 + 16 + l16] : 0.f;
#pragma unroll
  for (int r = 0; r < 8; ++r) {
    int row = m0 + r + 8 * hl;
    float v0 = acc0[r] + bc0;
    float v1 = acc1[r] + bc1;
    if (relu) { v0 = fmaxf(v0, 0.f); v1 = fmaxf(v1, 0.f); }
    size_t o = (size_t)row * N + n0 + l16;
    if (Cf) { Cf[o] = v0; Cf[o + 16] = v1; }
    if (Cb) { Cb[o] = (__bf16)v0; Cb[o + 16] = (__bf16)v1; }
  }
}

// ---------------------------------------------------------------------------
// 5) LayerNorm over E=128 (one wave per row), optional +pos, f32/bf16 out.
// ---------------------------------------------------------------------------
__global__ void __launch_bounds__(256) ln_kernel(
    const float* __restrict__ X, const float* __restrict__ pos,
    const float* __restrict__ g, const float* __restrict__ b,
    float* __restrict__ Yf, __bf16* __restrict__ Yb, int M) {
  int row = blockIdx.x * 8 + (threadIdx.x >> 5);
  int lane = threadIdx.x & 31;
  if (row >= M) return;
  int p = row & (PP - 1);
  float v[4];
  float s = 0.f;
#pragma unroll
  for (int i = 0; i < 4; ++i) {
    int e = lane + 32 * i;
    float t = X[(size_t)row * EE + e];
    if (pos) t += pos[(size_t)p * EE + e];
    v[i] = t; s += t;
  }
  for (int m = 1; m < 32; m <<= 1) s += __shfl_xor(s, m, 32);
  float mean = s * (1.f / 128.f);
  float var = 0.f;
#pragma unroll
  for (int i = 0; i < 4; ++i) { float d = v[i] - mean; var += d * d; }
  for (int m = 1; m < 32; m <<= 1) var += __shfl_xor(var, m, 32);
  float inv = rsqrtf(var * (1.f / 128.f) + 1e-5f);
#pragma unroll
  for (int i = 0; i < 4; ++i) {
    int e = lane + 32 * i;
    float y = (v[i] - mean) * inv * g[e] + b[e];
    if (Yf) Yf[(size_t)row * EE + e] = y;
    if (Yb) Yb[(size_t)row * EE + e] = (__bf16)y;
  }
}

// ---------------------------------------------------------------------------
// 6) qkv split: qh,kh [bh][P][32] bf16; V transposed to vt [bh][32][P].
// ---------------------------------------------------------------------------
__global__ void __launch_bounds__(256) qkv_split(const float* __restrict__ qkv,
                                                 __bf16* __restrict__ qh,
                                                 __bf16* __restrict__ kh,
                                                 __bf16* __restrict__ vt) {
  size_t i = (size_t)blockIdx.x * 256 + threadIdx.x;          // over 16384*384
  if (i >= (size_t)MM * 384) return;
  int m = (int)(i / 384), e = (int)(i % 384);
  int batch = m >> 10, p = m & (PP - 1);
  float val = qkv[i];
  int which = e >> 7, c = e & 127;
  int h = c >> 5, d = c & 31;
  size_t bh = (size_t)batch * 4 + h;
  if (which == 0)      qh[(bh * PP + p) * 32 + d] = (__bf16)val;
  else if (which == 1) kh[(bh * PP + p) * 32 + d] = (__bf16)val;
  else                 vt[(bh * 32 + d) * PP + p] = (__bf16)val;
}

// ---------------------------------------------------------------------------
// 7) Flash-style attention. One wave per (bh, 16-query tile).
//    2 score WMMAs per 32 keys (K=hd=32 exactly), online softmax with
//    __shfl_xor row reductions, LDS bounce C-layout -> A-fragment layout,
//    2 PV WMMAs against transposed V. Diagonal masked to -inf.
// ---------------------------------------------------------------------------
__global__ void __launch_bounds__(32) attn_kernel(const __bf16* __restrict__ qh,
                                                  const __bf16* __restrict__ kh,
                                                  const __bf16* __restrict__ vt,
                                                  __bf16* __restrict__ aout) {
  __shared__ __align__(32) __bf16 pl[16 * 32];   // prob tile [row][32 keys]
  const int bh = blockIdx.x;                     // 0..63
  const int batch = bh >> 2, head = bh & 3;
  const int q0 = blockIdx.y * 16;
  const int lane = threadIdx.x & 31;
  const int hl = lane >> 4, l16 = lane & 15;

  const __bf16* qb = qh + ((size_t)bh * PP + q0 + l16) * 32 + (hl ? 16 : 0);
  v16bf qf;
  ((v8bf*)&qf)[0] = *(const v8bf*)(qb);
  ((v8bf*)&qf)[1] = *(const v8bf*)(qb + 8);

  const __bf16* kb = kh + (size_t)bh * PP * 32;
  const __bf16* vb = vt + (size_t)bh * 32 * PP;

  v8f o0 = {}, o1 = {};
  float mrow[8], lrow[8];
#pragma unroll
  for (int r = 0; r < 8; ++r) { mrow[r] = -1e30f; lrow[r] = 0.f; }
  const float sc = 0.17677669529663687f;         // 1/sqrt(32)

  for (int kt = 0; kt < PP; kt += 32) {
    v16bf kf0 = *(const v16bf*)(kb + (size_t)(kt + l16) * 32 + (hl ? 16 : 0));
    v16bf kf1 = *(const v16bf*)(kb + (size_t)(kt + 16 + l16) * 32 + (hl ? 16 : 0));
    v8f s0 = {}, s1 = {};
    s0 = WMMA_BF16(qf, kf0, s0);
    s1 = WMMA_BF16(qf, kf1, s1);

#pragma unroll
    for (int r = 0; r < 8; ++r) {
      int qi = q0 + r + 8 * hl;
      s0[r] *= sc; s1[r] *= sc;
      if (qi == kt + l16)      s0[r] = -1e30f;   // mask diagonal
      if (qi == kt + 16 + l16) s1[r] = -1e30f;
      float tm = fmaxf(s0[r], s1[r]);
      tm = fmaxf(tm, __shfl_xor(tm, 1, 32));
      tm = fmaxf(tm, __shfl_xor(tm, 2, 32));
      tm = fmaxf(tm, __shfl_xor(tm, 4, 32));
      tm = fmaxf(tm, __shfl_xor(tm, 8, 32));
      float mn = fmaxf(mrow[r], tm);
      float corr = __expf(mrow[r] - mn);
      float p0 = __expf(s0[r] - mn);
      float p1 = __expf(s1[r] - mn);
      float ps = p0 + p1;
      ps += __shfl_xor(ps, 1, 32); ps += __shfl_xor(ps, 2, 32);
      ps += __shfl_xor(ps, 4, 32); ps += __shfl_xor(ps, 8, 32);
      lrow[r] = lrow[r] * corr + ps;
      mrow[r] = mn;
      o0[r] *= corr; o1[r] *= corr;
      int row = r + 8 * hl;
      pl[row * 32 + l16]      = (__bf16)p0;
      pl[row * 32 + 16 + l16] = (__bf16)p1;
    }
    __syncthreads();
    v16bf pf;
    ((v8bf*)&pf)[0] = *(const v8bf*)(pl + l16 * 32 + (hl ? 16 : 0));
    ((v8bf*)&pf)[1] = *(const v8bf*)(pl + l16 * 32 + 8 + (hl ? 16 : 0));
    v16bf vf0 = *(const v16bf*)(vb + (size_t)l16 * PP + kt + (hl ? 16 : 0));
    v16bf vf1 = *(const v16bf*)(vb + (size_t)(16 + l16) * PP + kt + (hl ? 16 : 0));
    o0 = WMMA_BF16(pf, vf0, o0);
    o1 = WMMA_BF16(pf, vf1, o1);
    __syncthreads();
  }
#pragma unroll
  for (int r = 0; r < 8; ++r) {
    int qi = q0 + r + 8 * hl;
    float inv = 1.f / lrow[r];
    size_t base = ((size_t)batch * PP + qi) * EE + head * 32;
    aout[base + l16]      = (__bf16)(o0[r] * inv);
    aout[base + 16 + l16] = (__bf16)(o1[r] * inv);
  }
}

// ---------------------------------------------------------------------------
// 8) Residual add -> bf16 (input to FFN GEMM).
// ---------------------------------------------------------------------------
__global__ void __launch_bounds__(256) resid_bf_kernel(const float* __restrict__ a,
                                                       const float* __restrict__ b,
                                                       __bf16* __restrict__ o, int n) {
  int i = blockIdx.x * 256 + threadIdx.x;
  if (i < n) o[i] = (__bf16)(a[i] + b[i]);
}

// ---------------------------------------------------------------------------
// 9) Build flow input rows (m = p*16 + batch): [x1(32) | pos(128) | g(128)].
// ---------------------------------------------------------------------------
__global__ void __launch_bounds__(256) build_flow(const float* __restrict__ z,
                                                  const float* __restrict__ pos,
                                                  const __bf16* __restrict__ g,
                                                  __bf16* __restrict__ Aflow) {
  size_t i = (size_t)blockIdx.x * 256 + threadIdx.x;          // over 16384*288
  if (i >= (size_t)MM * SUBIN) return;
  int m = (int)(i / SUBIN), c = (int)(i % SUBIN);
  int p = m >> 4, batch = m & 15;
  float v;
  if (c < SPLIT)       v = z[(size_t)m * CIN + c];
  else if (c < 160)    v = pos[(size_t)p * EE + (c - 32)];
  else                 v = (float)g[((size_t)batch * PP + p) * EE + (c - 160)];
  Aflow[i] = (__bf16)v;
}

// ---------------------------------------------------------------------------
// 10) Coupling law: s = 1.9*tanh(0.1*a[:32]); x2' = x2*exp(s) + 0.1*a[32:];
//     ldj += sum(s) + sum(log scale_i). One thread per row.
// ---------------------------------------------------------------------------
__global__ void __launch_bounds__(256) coupling(const float* __restrict__ a,
                                                const float* __restrict__ z,
                                                const float* __restrict__ gs,
                                                __bf16* __restrict__ zcb,
                                                float* __restrict__ ldj,
                                                int first_block) {
  int m = blockIdx.x * 256 + threadIdx.x;
  if (m >= MM) return;
  float acc = first_block ? 0.f : ldj[m];
  float slog = 0.f;
  for (int j = 0; j < CIN; ++j)
    slog += __logf(0.2f * log1pf(__expf(0.5f * gs[j])));
  for (int j = 0; j < SPLIT; ++j) {
    float s  = 1.9f * tanhf(0.1f * a[(size_t)m * CIN + j]);
    float t  = 0.1f * a[(size_t)m * CIN + SPLIT + j];
    float x2 = z[(size_t)m * CIN + SPLIT + j] * __expf(s) + t;
    acc += s;
    zcb[(size_t)m * CIN + j]         = (__bf16)z[(size_t)m * CIN + j];
    zcb[(size_t)m * CIN + SPLIT + j] = (__bf16)x2;
  }
  ldj[m] = acc + slog;
}

// ---------------------------------------------------------------------------
// 11) z = (zc @ w_perm^T) * scale + offset; also refresh x1 in Aflow.
// ---------------------------------------------------------------------------
__global__ void __launch_bounds__(256) perm_scale(const float* __restrict__ zperm,
                                                  const float* __restrict__ gs,
                                                  const float* __restrict__ go,
                                                  float* __restrict__ znew,
                                                  __bf16* __restrict__ Aflow) {
  size_t i = (size_t)blockIdx.x * 256 + threadIdx.x;          // over 16384*64
  if (i >= (size_t)MM * CIN) return;
  int m = (int)(i >> 6), j = (int)(i & 63);
  float sc = 0.2f * log1pf(__expf(0.5f * gs[j]));
  float v = zperm[i] * sc + go[j];
  znew[i] = v;
  if (Aflow && j < SPLIT) Aflow[(size_t)m * SUBIN + j] = (__bf16)v;
}

// ---------------------------------------------------------------------------
// 12) Pack outputs: zs (P,b,c) flat == z rows; jacs appended.
// ---------------------------------------------------------------------------
__global__ void __launch_bounds__(256) finalize(const float* __restrict__ z,
                                                const float* __restrict__ ldj,
                                                float* __restrict__ out) {
  size_t i = (size_t)blockIdx.x * 256 + threadIdx.x;
  if (i < (size_t)MM * CIN) out[i] = z[i];
  else if (i < (size_t)MM * CIN + MM) out[i] = ldj[i - (size_t)MM * CIN];
}

// ---------------------------------------------------------------------------
// Host orchestration.
// ---------------------------------------------------------------------------
static inline void launch_gemm(const __bf16* A, const __bf16* W, const float* bias,
                               float* Cf, __bf16* Cb, int M, int N, int K, int relu,
                               hipStream_t s) {
  int blocks = (M / 128) * (N / 32);
  size_t lds = (size_t)(K / 32) * 2048;          // (K/32)*2 slots * 1KB
  gemm_bf16_wmma<<<blocks, 256, lds, s>>>(A, W, bias, Cf, Cb, M, N, K, relu);
}

extern "C" void kernel_launch(void* const* d_in, const int* in_sizes, int n_in,
                              void* d_out, int out_size, void* d_ws, size_t ws_size,
                              hipStream_t stream) {
  (void)in_sizes; (void)n_in; (void)out_size; (void)ws_size;
  const float* x      = (const float*)d_in[0];
  const float* proj_w = (const float*)d_in[1];
  const float* proj_b = (const float*)d_in[2];
  const float* ln1_g  = (const float*)d_in[3];
  const float* ln1_b  = (const float*)d_in[4];
  const float* in_w   = (const float*)d_in[5];
  const float* in_b   = (const float*)d_in[6];
  const float* out_w  = (const float*)d_in[7];
  const float* out_b  = (const float*)d_in[8];
  const float* ffn_w  = (const float*)d_in[9];
  const float* ffn_b  = (const float*)d_in[10];
  const float* ln2_g  = (const float*)d_in[11];
  const float* ln2_b  = (const float*)d_in[12];
  const float* sub_w1 = (const float*)d_in[13];
  const float* sub_b1 = (const float*)d_in[14];
  const float* sub_w2 = (const float*)d_in[15];
  const float* sub_b2 = (const float*)d_in[16];
  const float* w_perm = (const float*)d_in[17];
  const float* gsc    = (const float*)d_in[18];
  const float* gof    = (const float*)d_in[19];

  char* base = (char*)d_ws;
  size_t off = 0;
  auto alloc = [&](size_t bytes) -> void* {
    void* p = base + off;
    off = (off + bytes + 255) & ~(size_t)255;
    return p;
  };

  __bf16* wb_proj = (__bf16*)alloc((size_t)EE * CIN * 2);
  __bf16* wb_in   = (__bf16*)alloc((size_t)3 * EE * EE * 2);
  __bf16* wb_out  = (__bf16*)alloc((size_t)EE * EE * 2);
  __bf16* wb_ffn  = (__bf16*)alloc((size_t)EE * EE * 2);
  __bf16* wb_s1   = (__bf16*)alloc((size_t)2 * SUBHID * SUBIN * 2);
  __bf16* wb_s2   = (__bf16*)alloc((size_t)2 * CIN * SUBHID * 2);
  __bf16* wb_pm   = (__bf16*)alloc((size_t)2 * CIN * CIN * 2);
  float*  pos     = (float*) alloc((size_t)PP * EE * 4);
  float*  z0      = (float*) alloc((size_t)MM * CIN * 4);
  __bf16* A1      = (__bf16*)alloc((size_t)MM * CIN * 2);
  float*  hh_pre  = (float*) alloc((size_t)MM * EE * 4);
  float*  hh      = (float*) alloc((size_t)MM * EE * 4);
  __bf16* hh_bf   = (__bf16*)alloc((size_t)MM * EE * 2);
  float*  qkv     = (float*) alloc((size_t)MM * 384 * 4);
  __bf16* qh      = (__bf16*)alloc((size_t)64 * PP * 32 * 2);
  __bf16* kh      = (__bf16*)alloc((size_t)64 * PP * 32 * 2);
  __bf16* vt      = (__bf16*)alloc((size_t)64 * 32 * PP * 2);
  __bf16* aot     = (__bf16*)alloc((size_t)MM * EE * 2);
  float*  opre    = (float*) alloc((size_t)MM * EE * 4);
  __bf16* rbf     = (__bf16*)alloc((size_t)MM * EE * 2);
  float*  gpre    = (float*) alloc((size_t)MM * EE * 4);
  __bf16* gbf     = (__bf16*)alloc((size_t)MM * EE * 2);
  __bf16* Aflow   = (__bf16*)alloc((size_t)MM * SUBIN * 2);
  __bf16* h1bf    = (__bf16*)alloc((size_t)MM * SUBHID * 2);
  float*  aoutb   = (float*) alloc((size_t)MM * CIN * 4);
  __bf16* zcb     = (__bf16*)alloc((size_t)MM * CIN * 2);
  float*  zperm   = (float*) alloc((size_t)MM * CIN * 4);
  float*  z1      = (float*) alloc((size_t)MM * CIN * 4);
  float*  ldj     = (float*) alloc((size_t)MM * 4);

  // ---- Stage 0: pooling, positional encoding, weight conversion ----
  pool_kernel<<<4096, 256, 0, stream>>>(x, z0, A1);
  pos_kernel<<<512, 256, 0, stream>>>(pos);
  f2b<<<(EE * CIN + 255) / 256, 256, 0, stream>>>(proj_w, wb_proj, EE * CIN);
  f2b<<<(3 * EE * EE + 255) / 256, 256, 0, stream>>>(in_w, wb_in, 3 * EE * EE);
  f2b<<<(EE * EE + 255) / 256, 256, 0, stream>>>(out_w, wb_out, EE * EE);
  f2b<<<(EE * EE + 255) / 256, 256, 0, stream>>>(ffn_w, wb_ffn, EE * EE);
  f2b<<<(2 * SUBHID * SUBIN + 255) / 256, 256, 0, stream>>>(sub_w1, wb_s1, 2 * SUBHID * SUBIN);
  f2b<<<(2 * CIN * SUBHID + 255) / 256, 256, 0, stream>>>(sub_w2, wb_s2, 2 * CIN * SUBHID);
  f2b<<<(2 * CIN * CIN + 255) / 256, 256, 0, stream>>>(w_perm, wb_pm, 2 * CIN * CIN);

  // ---- Stage 1: transformer ----
  launch_gemm(A1, wb_proj, proj_b, hh_pre, nullptr, MM, EE, CIN, 0, stream);
  ln_kernel<<<MM / 8, 256, 0, stream>>>(hh_pre, pos, ln1_g, ln1_b, hh, hh_bf, MM);
  launch_gemm(hh_bf, wb_in, in_b, qkv, nullptr, MM, 3 * EE, EE, 0, stream);
  qkv_split<<<(int)(((size_t)MM * 384 + 255) / 256), 256, 0, stream>>>(qkv, qh, kh, vt);
  attn_kernel<<<dim3(64, 64), 32, 0, stream>>>(qh, kh, vt, aot);
  launch_gemm(aot, wb_out, out_b, opre, nullptr, MM, EE, EE, 0, stream);
  resid_bf_kernel<<<(MM * EE) / 256, 256, 0, stream>>>(hh, opre, rbf, MM * EE);
  launch_gemm(rbf, wb_ffn, ffn_b, gpre, nullptr, MM, EE, EE, 0, stream);
  ln_kernel<<<MM / 8, 256, 0, stream>>>(gpre, nullptr, ln2_g, ln2_b, nullptr, gbf, MM);

  // ---- Stage 2: normalizing-flow blocks ----
  build_flow<<<(int)(((size_t)MM * SUBIN + 255) / 256), 256, 0, stream>>>(z0, pos, gbf, Aflow);
  float* zcur = z0;
  float* znext = z1;
  for (int i = 0; i < 2; ++i) {
    launch_gemm(Aflow, wb_s1 + (size_t)i * SUBHID * SUBIN, sub_b1 + i * SUBHID,
                nullptr, h1bf, MM, SUBHID, SUBIN, 1, stream);
    launch_gemm(h1bf, wb_s2 + (size_t)i * CIN * SUBHID, sub_b2 + i * CIN,
                aoutb, nullptr, MM, CIN, SUBHID, 0, stream);
    coupling<<<MM / 256, 256, 0, stream>>>(aoutb, zcur, gsc + i * CIN, zcb, ldj, i == 0);
    launch_gemm(zcb, wb_pm + (size_t)i * CIN * CIN, nullptr, zperm, nullptr,
                MM, CIN, CIN, 0, stream);
    perm_scale<<<(MM * CIN) / 256, 256, 0, stream>>>(zperm, gsc + i * CIN, gof + i * CIN,
                                                     znext, (i == 0) ? Aflow : nullptr);
    float* t = zcur; zcur = znext; znext = t;
  }
  finalize<<<(int)(((size_t)MM * CIN + MM + 255) / 256), 256, 0, stream>>>(zcur, ldj, (float*)d_out);
}
